// GatedCrossAttn_38010460570002
// MI455X (gfx1250) — matrix operand order repaired and checked by reading, not compiled, not run on hardware
//
#include <hip/hip_runtime.h>
#include <math.h>

// ---------------------------------------------------------------------------
// Problem constants (from reference)
// ---------------------------------------------------------------------------
#define Dm    1024
#define Hn    16
#define DK    64
#define Bb    8
#define NX    512
#define NY    1024
#define LAMg  2.0f
#define GMINg 0.05f
#define EPSg  1e-6f
#define SCALEa 0.125f   // 1/sqrt(64)

typedef __bf16 v16bf __attribute__((ext_vector_type(16)));
typedef float  v8f   __attribute__((ext_vector_type(8)));

// --- optional CDNA5 async global->LDS copy path (ASYNCcnt tracked) ---------
#if defined(__has_builtin)
#if __has_builtin(__builtin_amdgcn_global_load_async_to_lds_b128) && \
    __has_builtin(__builtin_amdgcn_s_wait_asynccnt)
#define HAVE_ASYNC_LDS 1
#endif
#endif

#if defined(HAVE_ASYNC_LDS)
// builtin signature (from hipcc diagnostic): (int4 AS1*, int4 AS3*, imm, imm)
typedef int v4i_vs __attribute__((vector_size(16)));
typedef __attribute__((address_space(1))) v4i_vs* glb_v4i_p;
typedef __attribute__((address_space(3))) v4i_vs* lds_v4i_p;
#define ASYNC_CP16(gsrc, ldst)                                              \
  __builtin_amdgcn_global_load_async_to_lds_b128(                           \
      (glb_v4i_p)(size_t)(gsrc), (lds_v4i_p)(ldst), 0, 0)
#endif

static __device__ __forceinline__ float redsum16(float v) {
#pragma unroll
  for (int m = 1; m < 16; m <<= 1) v += __shfl_xor(v, m, 16);
  return v;
}
static __device__ __forceinline__ float redmax16(float v) {
#pragma unroll
  for (int m = 1; m < 16; m <<= 1) v = fmaxf(v, __shfl_xor(v, m, 16));
  return v;
}
static __device__ __forceinline__ float redsum32(float v) {
#pragma unroll
  for (int m = 1; m < 32; m <<= 1) v += __shfl_xor(v, m, 32);
  return v;
}
static __device__ __forceinline__ float softplusf(float x) {
  return (x > 20.f) ? x : log1pf(expf(x));
}

// ---------------------------------------------------------------------------
// fp32 -> bf16 conversion (n multiple of 1024)
// ---------------------------------------------------------------------------
__global__ __launch_bounds__(256)
void cvt_bf16_kernel(const float* __restrict__ src, __bf16* __restrict__ dst) {
  const size_t i = ((size_t)blockIdx.x * 256 + threadIdx.x) * 4;
  const float4 v = *(const float4*)&src[i];
  dst[i + 0] = (__bf16)v.x; dst[i + 1] = (__bf16)v.y;
  dst[i + 2] = (__bf16)v.z; dst[i + 3] = (__bf16)v.w;
}

// ---------------------------------------------------------------------------
// Evidential gate: one wave per token.  g -> gout, log(g) -> loggout
// ---------------------------------------------------------------------------
__global__ __launch_bounds__(256)
void gate_kernel(const float* __restrict__ X, const float* __restrict__ gw,
                 const float* __restrict__ gb, float* __restrict__ gout,
                 float* __restrict__ loggout, int T) {
  const int lane = threadIdx.x & 31, wave = threadIdx.x >> 5;
  const int t = blockIdx.x * 8 + wave;
  if (t >= T) return;
  const float* xr = X + (size_t)t * Dm;
  float a0 = 0.f, a1 = 0.f, a2 = 0.f, a3 = 0.f;
  for (int d = lane; d < Dm; d += 32) {
    const float xv = xr[d];
    a0 += xv * gw[d];
    a1 += xv * gw[Dm + d];
    a2 += xv * gw[2 * Dm + d];
    a3 += xv * gw[3 * Dm + d];
  }
  a0 = redsum32(a0); a1 = redsum32(a1); a2 = redsum32(a2); a3 = redsum32(a3);
  if (lane == 0) {
    const float mu    = a0 + gb[0];
    const float v     = softplusf(a1 + gb[1]) + EPSg;
    const float alpha = softplusf(a2 + gb[2]) + 1.0f + EPSg;
    const float beta  = softplusf(a3 + gb[3]) + EPSg;
    const float var   = beta / (v * (alpha - 1.0f));
    float g = (1.f / (1.f + expf(-mu))) * expf(-LAMg * fmaxf(var, 0.f));
    g = fminf(fmaxf(g, GMINg), 1.0f);
    g = fmaxf(g, 1e-6f);
    gout[t]    = g;
    loggout[t] = logf(g);
  }
}

// ---------------------------------------------------------------------------
// bf16 WMMA GEMM:  C[M,N] = oscale * act( A[M,K] * W[N,K]^T + bias )
// Block tile 128x128 (4 waves), wave tile 64x64: 16 v_wmma per K=32 step.
// Double-buffered LDS with ASYNCcnt-pipelined global->LDS copies:
// group n+1 is issued before computing group n; s_wait_asynccnt(8) retires
// only the older 8 async ops per wave (in-order completion).
// ---------------------------------------------------------------------------
__global__ __launch_bounds__(128)
void gemm_bf16_kernel(const __bf16* __restrict__ A, const __bf16* __restrict__ W,
                      const float* __restrict__ bias, float* __restrict__ Cf,
                      __bf16* __restrict__ Cb, int M, int N, int K, int act,
                      float oscale) {
  __shared__ __bf16 As[2][128][40];   // padded rows: 80B stride
  __shared__ __bf16 Bs[2][128][40];
  const int tid  = threadIdx.x;
  const int lane = tid & 31, wave = tid >> 5;
  const int l15  = lane & 15, hf = lane >> 4;
  const int wr = wave >> 1, wc = wave & 1;          // wave grid 2x2
  const int m0 = blockIdx.y * 128, n0 = blockIdx.x * 128;
  const int kbA = hf * 8;    // A-frag K base (documented 16-bit A layout)
  const int kbB = hf * 8;    // B-frag K base (documented 16-bit B layout)

  v8f acc[4][4];
#pragma unroll
  for (int a = 0; a < 4; ++a)
#pragma unroll
    for (int c = 0; c < 4; ++c)
      acc[a][c] = (v8f){0.f,0.f,0.f,0.f,0.f,0.f,0.f,0.f};

  // stage one 128x32 K-slice of A and W into LDS buffer `buf`
  auto stage = [&](int kk, int buf) {
#pragma unroll
    for (int i = 0; i < 4; ++i) {
      const int slot = tid + i * 128;        // 0..511
      const int r    = slot >> 2;            // 0..127
      const int c8   = (slot & 3) * 8;       // 0..24
      const __bf16* ga = &A[(size_t)(m0 + r) * K + kk + c8];
      const __bf16* gb = &W[(size_t)(n0 + r) * K + kk + c8];
#if defined(HAVE_ASYNC_LDS)
      ASYNC_CP16(ga, &As[buf][r][c8]);
      ASYNC_CP16(gb, &Bs[buf][r][c8]);
#else
      *(uint4*)&As[buf][r][c8] = *(const uint4*)ga;
      *(uint4*)&Bs[buf][r][c8] = *(const uint4*)gb;
#endif
    }
  };

  const int nk = K >> 5;
  stage(0, 0);
  for (int n = 0; n < nk; ++n) {
    const int cur = n & 1;
    if (n + 1 < nk) {
      stage((n + 1) << 5, cur ^ 1);          // prefetch next slice
#if defined(HAVE_ASYNC_LDS)
      __builtin_amdgcn_s_wait_asynccnt(8);   // older group landed, newer in flight
#endif
    } else {
#if defined(HAVE_ASYNC_LDS)
      __builtin_amdgcn_s_wait_asynccnt(0);
#endif
    }
    __syncthreads();                         // all waves' slice-n writes visible

    // fragments: 4 A (rows) x 4 B (cols) -> 16 wmma
    v16bf af[4], bf[4];
#pragma unroll
    for (int t = 0; t < 4; ++t) {
      const __bf16* ar = &As[cur][wr * 64 + t * 16 + l15][0];
#pragma unroll
      for (int j = 0; j < 8; ++j) {
        af[t][j]     = ar[kbA + j];
        af[t][8 + j] = ar[kbA + 16 + j];
      }
      const __bf16* br = &Bs[cur][wc * 64 + t * 16 + l15][kbB];
#pragma unroll
      for (int j = 0; j < 16; ++j) bf[t][j] = br[j + (j & 8)];
    }
#pragma unroll
    for (int a = 0; a < 4; ++a)
#pragma unroll
      for (int c = 0; c < 4; ++c)
        acc[a][c] = __builtin_amdgcn_wmma_f32_16x16x32_bf16(
            false, af[a], false, bf[c], (short)0, acc[a][c], false, false);

    __syncthreads();   // protect buffer `cur` before it is refilled at n+2
  }

  // epilogue
#pragma unroll
  for (int a = 0; a < 4; ++a) {
#pragma unroll
    for (int c = 0; c < 4; ++c) {
      const int n = n0 + wc * 64 + c * 16 + l15;
      const float bv = bias ? bias[n] : 0.f;
#pragma unroll
      for (int r = 0; r < 8; ++r) {
        const int m = m0 + wr * 64 + a * 16 + r + hf * 8;
        float v = acc[a][c][r] + bv;
        if (act == 1) v = v / (1.f + expf(-v));   // SiLU
        v *= oscale;
        if (Cb) Cb[(size_t)m * N + n] = (__bf16)v;
        else    Cf[(size_t)m * N + n] = v;
      }
    }
  }
}

// ---------------------------------------------------------------------------
// Gated cross attention (flash-style, online softmax), bf16 in / bf16 out.
// Q already scaled by 1/sqrt(dk) (folded into projection GEMM).
// grid = (Tq/64, H, B), 128 threads = 4 waves; wave owns 16 q rows.
// ---------------------------------------------------------------------------
__global__ __launch_bounds__(128)
void attn_kernel(const __bf16* __restrict__ Q, const __bf16* __restrict__ Kb,
                 const __bf16* __restrict__ Vb, const float* __restrict__ logg,
                 __bf16* __restrict__ O, int Tq, int Tk) {
  __shared__ __bf16 Kt[32][72];      // [key][dk], padded
  __shared__ __bf16 Vt[64][40];      // [dk][key], padded (transposed)
  __shared__ __bf16 Ps[4][16][40];   // per-wave P tile (C-layout -> A-layout)

  const int tid  = threadIdx.x;
  const int lane = tid & 31, wave = tid >> 5;
  const int l15  = lane & 15, hf = lane >> 4;
  const int b = blockIdx.z, h = blockIdx.y;
  const int q0  = blockIdx.x * 64 + wave * 16;
  const int dk0 = h * DK;
  const int kbA = hf * 8;
  const int kbB = hf * 8;

  // Q fragments (two K=32 chunks over DK=64)
  const __bf16* qrow = Q + ((size_t)(b * Tq) + q0 + l15) * Dm + dk0;
  v16bf qa[2];
#pragma unroll
  for (int ck = 0; ck < 2; ++ck) {
#pragma unroll
    for (int j = 0; j < 8; ++j) {
      qa[ck][j]     = qrow[ck * 32 + kbA + j];
      qa[ck][8 + j] = qrow[ck * 32 + kbA + 16 + j];
    }
  }

  float mrow[8], lrow[8];
  v8f o[4];
#pragma unroll
  for (int r = 0; r < 8; ++r) { mrow[r] = -3.0e38f; lrow[r] = 0.f; }
#pragma unroll
  for (int c = 0; c < 4; ++c) o[c] = (v8f){0.f,0.f,0.f,0.f,0.f,0.f,0.f,0.f};

  const __bf16* kbase = Kb + (size_t)(b * Tk) * Dm + dk0;
  const __bf16* vbase = Vb + (size_t)(b * Tk) * Dm + dk0;
  const float*  gbase = logg + b * Tk;

  for (int kb = 0; kb < Tk; kb += 32) {
    __syncthreads();   // previous iteration done reading Kt/Vt
#pragma unroll
    for (int i = 0; i < 2; ++i) {
      const int slot = tid + i * 128;      // 0..255 (8-elem chunks)
      const int key  = slot >> 3;          // 0..31
      const int d8   = (slot & 7) * 8;     // 0..56
      const __bf16* kr = &kbase[(size_t)(kb + key) * Dm + d8];
#if defined(HAVE_ASYNC_LDS)
      ASYNC_CP16(kr, &Kt[key][d8]);
#else
      *(uint4*)&Kt[key][d8] = *(const uint4*)kr;
#endif
      const __bf16* vr = &vbase[(size_t)(kb + key) * Dm + d8];
#pragma unroll
      for (int j = 0; j < 8; ++j) Vt[d8 + j][key] = vr[j];  // transpose
    }
#if defined(HAVE_ASYNC_LDS)
    __builtin_amdgcn_s_wait_asynccnt(0);
#endif
    __syncthreads();

    // --- scores: two 16x16 tiles covering 32 keys ---
    v8f s[2];
#pragma unroll
    for (int g = 0; g < 2; ++g) {
      s[g] = (v8f){0.f,0.f,0.f,0.f,0.f,0.f,0.f,0.f};
#pragma unroll
      for (int ck = 0; ck < 2; ++ck) {
        const __bf16* krow = &Kt[g * 16 + l15][ck * 32 + kbB];
        v16bf kf;
#pragma unroll
        for (int j = 0; j < 16; ++j) kf[j] = krow[j + (j & 8)];
        s[g] = __builtin_amdgcn_wmma_f32_16x16x32_bf16(
            false, qa[ck], false, kf, (short)0, s[g], false, false);
      }
      const float lg = gbase[kb + g * 16 + l15];   // per-key log-gate bias
#pragma unroll
      for (int r = 0; r < 8; ++r) s[g][r] += lg;
    }

    // --- online softmax update (rows live in 16-lane halves) ---
#pragma unroll
    for (int r = 0; r < 8; ++r) {
      const float cm  = redmax16(fmaxf(s[0][r], s[1][r]));
      const float nm  = fmaxf(mrow[r], cm);
      const float fac = expf(mrow[r] - nm);
      mrow[r] = nm;
      const float p0 = expf(s[0][r] - nm);
      const float p1 = expf(s[1][r] - nm);
      lrow[r] = lrow[r] * fac + redsum16(p0 + p1);
#pragma unroll
      for (int c = 0; c < 4; ++c) o[c][r] *= fac;
      const int row = r + hf * 8;
      Ps[wave][row][l15]      = (__bf16)p0;
      Ps[wave][row][16 + l15] = (__bf16)p1;
    }
    __syncthreads();   // P tiles visible (keeps waves in lockstep too)

    // --- P (16x32) x V (32x64) accumulate into O ---
    const __bf16* prow = &Ps[wave][l15][0];
    v16bf pf;
#pragma unroll
    for (int j = 0; j < 8; ++j) {
      pf[j]     = prow[kbA + j];
      pf[8 + j] = prow[kbA + 16 + j];
    }
#pragma unroll
    for (int c = 0; c < 4; ++c) {
      const __bf16* vrow = &Vt[c * 16 + l15][kbB];
      v16bf vf;
#pragma unroll
      for (int j = 0; j < 16; ++j) vf[j] = vrow[j + (j & 8)];
      o[c] = __builtin_amdgcn_wmma_f32_16x16x32_bf16(
          false, pf, false, vf, (short)0, o[c], false, false);
    }
  }

  // --- epilogue: normalize, write merged bf16 layout [token, h*64+d] ---
#pragma unroll
  for (int c = 0; c < 4; ++c) {
#pragma unroll
    for (int r = 0; r < 8; ++r) {
      const int row = r + hf * 8;
      O[((size_t)(b * Tq) + q0 + row) * Dm + dk0 + c * 16 + l15] =
          (__bf16)(o[c][r] / lrow[r]);
    }
  }
}

// ---------------------------------------------------------------------------
// out = LayerNorm(a + bsrc) * g + beta  (row length 1024, one block per row)
// Optionally also writes a bf16 copy (for feeding the next WMMA GEMM).
// ---------------------------------------------------------------------------
__global__ __launch_bounds__(256)
void add_ln_kernel(const float* __restrict__ a, const float* __restrict__ bsrc,
                   const float* __restrict__ g, const float* __restrict__ beta,
                   float* __restrict__ out, __bf16* __restrict__ out_bf) {
  __shared__ float red[2][8];
  const size_t row = blockIdx.x;
  const int tid = threadIdx.x, lane = tid & 31, wave = tid >> 5;
  float v[4];
  float s = 0.f, s2 = 0.f;
#pragma unroll
  for (int i = 0; i < 4; ++i) {
    const int c = tid + i * 256;
    v[i] = a[row * Dm + c] + bsrc[row * Dm + c];
    s  += v[i];
    s2 += v[i] * v[i];
  }
  s = redsum32(s); s2 = redsum32(s2);
  if (lane == 0) { red[0][wave] = s; red[1][wave] = s2; }
  __syncthreads();
  float ts = 0.f, ts2 = 0.f;
#pragma unroll
  for (int w = 0; w < 8; ++w) { ts += red[0][w]; ts2 += red[1][w]; }
  const float mu  = ts * (1.f / Dm);
  const float var = ts2 * (1.f / Dm) - mu * mu;
  const float ri  = rsqrtf(var + 1e-5f);
#pragma unroll
  for (int i = 0; i < 4; ++i) {
    const int c = tid + i * 256;
    const float r = (v[i] - mu) * ri * g[c] + beta[c];
    out[row * Dm + c] = r;
    if (out_bf) out_bf[row * Dm + c] = (__bf16)r;
  }
}

// ---------------------------------------------------------------------------
// Host-side orchestration
// ---------------------------------------------------------------------------
extern "C" void kernel_launch(void* const* d_in, const int* in_sizes, int n_in,
                              void* d_out, int out_size, void* d_ws, size_t ws_size,
                              hipStream_t stream) {
  (void)in_sizes; (void)n_in; (void)out_size; (void)ws_size;
  const float* x     = (const float*)d_in[0];
  const float* y     = (const float*)d_in[1];
  const float* Wqx   = (const float*)d_in[4];
  const float* Wkx   = (const float*)d_in[5];
  const float* Wvx   = (const float*)d_in[6];
  const float* Wqy   = (const float*)d_in[7];
  const float* Wky   = (const float*)d_in[8];
  const float* Wvy   = (const float*)d_in[9];
  const float* gx_w  = (const float*)d_in[10];
  const float* gx_b  = (const float*)d_in[11];
  const float* gy_w  = (const float*)d_in[12];
  const float* gy_b  = (const float*)d_in[13];
  const float* Wox   = (const float*)d_in[14];
  const float* Woy   = (const float*)d_in[15];
  const float* lnx_g = (const float*)d_in[16];
  const float* lnx_b = (const float*)d_in[17];
  const float* lny_g = (const float*)d_in[18];
  const float* lny_b = (const float*)d_in[19];
  const float* ffx_w = (const float*)d_in[20];
  const float* ffx_b = (const float*)d_in[21];
  const float* ffy_w = (const float*)d_in[22];
  const float* ffy_b = (const float*)d_in[23];

  const int NXT = Bb * NX;   // 4096 x-tokens
  const int NYT = Bb * NY;   // 8192 y-tokens
  const size_t SX = (size_t)NXT * Dm;   // 4,194,304
  const size_t SY = (size_t)NYT * Dm;   // 8,388,608
  const size_t SW = (size_t)Dm * Dm;    // 1,048,576

  float* out_x2 = (float*)d_out;
  float* out_y2 = out_x2 + SX;
  float* out_gx = out_y2 + SY;
  float* out_gy = out_gx + NXT;

  // workspace carve (bytes, 256B aligned per buffer)
  char* wsb = (char*)d_ws;
  size_t off = 0;
  auto carve = [&](size_t bytes) {
    char* p = wsb + off;
    off += (bytes + 255) & ~(size_t)255;
    return (void*)p;
  };
  __bf16* x_bf   = (__bf16*)carve(SX * 2);
  __bf16* y_bf   = (__bf16*)carve(SY * 2);
  __bf16* wbf[10];
  for (int i = 0; i < 10; ++i) wbf[i] = (__bf16*)carve(SW * 2);
  __bf16* qx_bf  = (__bf16*)carve(SX * 2);
  __bf16* kx_bf  = (__bf16*)carve(SX * 2);
  __bf16* vx_bf  = (__bf16*)carve(SX * 2);
  __bf16* qy_bf  = (__bf16*)carve(SY * 2);
  __bf16* ky_bf  = (__bf16*)carve(SY * 2);
  __bf16* vy_bf  = (__bf16*)carve(SY * 2);
  float*  tx     = (float*)carve(SX * 4);
  float*  ty     = (float*)carve(SY * 4);
  float*  x2a    = (float*)carve(SX * 4);
  float*  y2a    = (float*)carve(SY * 4);
  float*  loggx  = (float*)carve(NXT * 4);
  float*  loggy  = (float*)carve(NYT * 4);
  // reuse buffers whose lifetimes ended:
  __bf16* ax_bf  = x_bf;    // x_bf dead after QKV projections
  __bf16* ay_bf  = y_bf;
  __bf16* x2a_bf = qx_bf;   // qx_bf dead after attention x->y
  __bf16* y2a_bf = qy_bf;
  float*  txf    = tx;      // tx dead after first add_ln
  float*  tyf    = ty;

  const float* wsrc[10] = {Wqx, Wkx, Wvx, Wqy, Wky, Wvy, Wox, Woy, ffx_w, ffy_w};

  const dim3 blk128(128), blk256(256);

  // 0) one-time fp32 -> bf16 conversions (activations + all weights)
  cvt_bf16_kernel<<<dim3(SX / 1024), blk256, 0, stream>>>(x, x_bf);
  cvt_bf16_kernel<<<dim3(SY / 1024), blk256, 0, stream>>>(y, y_bf);
  for (int i = 0; i < 10; ++i)
    cvt_bf16_kernel<<<dim3(SW / 1024), blk256, 0, stream>>>(wsrc[i], wbf[i]);

  // 1) evidential gates (g -> d_out, log g -> ws)
  gate_kernel<<<dim3(NXT / 8), blk256, 0, stream>>>(x, gx_w, gx_b, out_gx, loggx, NXT);
  gate_kernel<<<dim3(NYT / 8), blk256, 0, stream>>>(y, gy_w, gy_b, out_gy, loggy, NYT);

  // 2) QKV projections (bf16 WMMA, bf16 out; SCALE folded into Q)
  const dim3 ggx(Dm / 128, NXT / 128);   // (8, 32)
  const dim3 ggy(Dm / 128, NYT / 128);   // (8, 64)
  gemm_bf16_kernel<<<ggx, blk128, 0, stream>>>(x_bf, wbf[0], nullptr, nullptr, qx_bf, NXT, Dm, Dm, 0, SCALEa);
  gemm_bf16_kernel<<<ggx, blk128, 0, stream>>>(x_bf, wbf[1], nullptr, nullptr, kx_bf, NXT, Dm, Dm, 0, 1.f);
  gemm_bf16_kernel<<<ggx, blk128, 0, stream>>>(x_bf, wbf[2], nullptr, nullptr, vx_bf, NXT, Dm, Dm, 0, 1.f);
  gemm_bf16_kernel<<<ggy, blk128, 0, stream>>>(y_bf, wbf[3], nullptr, nullptr, qy_bf, NYT, Dm, Dm, 0, SCALEa);
  gemm_bf16_kernel<<<ggy, blk128, 0, stream>>>(y_bf, wbf[4], nullptr, nullptr, ky_bf, NYT, Dm, Dm, 0, 1.f);
  gemm_bf16_kernel<<<ggy, blk128, 0, stream>>>(y_bf, wbf[5], nullptr, nullptr, vy_bf, NYT, Dm, Dm, 0, 1.f);

  // 3) cross attentions (per-key log-gate bias; per-query bias cancels)
  attn_kernel<<<dim3(NX / 64, Hn, Bb), blk128, 0, stream>>>(qx_bf, ky_bf, vy_bf, loggy, ax_bf, NX, NY);
  attn_kernel<<<dim3(NY / 64, Hn, Bb), blk128, 0, stream>>>(qy_bf, kx_bf, vx_bf, loggx, ay_bf, NY, NX);

  // 4) output projections (fp32 out for LN precision)
  gemm_bf16_kernel<<<ggx, blk128, 0, stream>>>(ax_bf, wbf[6], nullptr, tx, nullptr, NXT, Dm, Dm, 0, 1.f);
  gemm_bf16_kernel<<<ggy, blk128, 0, stream>>>(ay_bf, wbf[7], nullptr, ty, nullptr, NYT, Dm, Dm, 0, 1.f);

  // 5) residual + LN (fp32 + bf16 copy for the FFN GEMM)
  add_ln_kernel<<<dim3(NXT), blk256, 0, stream>>>(x, tx, lnx_g, lnx_b, x2a, x2a_bf);
  add_ln_kernel<<<dim3(NYT), blk256, 0, stream>>>(y, ty, lny_g, lny_b, y2a, y2a_bf);

  // 6) FFN with fused SiLU (fp32 out)
  gemm_bf16_kernel<<<ggx, blk128, 0, stream>>>(x2a_bf, wbf[8], ffx_b, txf, nullptr, NXT, Dm, Dm, 1, 1.f);
  gemm_bf16_kernel<<<ggy, blk128, 0, stream>>>(y2a_bf, wbf[9], ffy_b, tyf, nullptr, NYT, Dm, Dm, 1, 1.f);

  // 7) residual + LN -> final outputs
  add_ln_kernel<<<dim3(NXT), blk256, 0, stream>>>(x2a, txf, lnx_g, lnx_b, out_x2, nullptr);
  add_ln_kernel<<<dim3(NYT), blk256, 0, stream>>>(y2a, tyf, lny_g, lny_b, out_y2, nullptr);
}
